// TimeAwareCondConv2d_73469710565801
// MI455X (gfx1250) — matrix operand order repaired and checked by reading, not compiled, not run on hardware
//
#include <hip/hip_runtime.h>
#include <stdint.h>

// ---------------- problem constants ----------------
#define BB 16
#define CC 256
#define OO 256
#define HH 64
#define WW 64
#define TDIM 512
#define EE 4
#define HW 4096           // H*W

// ---------------- workspace layout (bytes) ----------------
#define WS_POOLED 0                       // f32 [16][256]      = 16384
#define WS_RW     16384                   // f32 [16][4]        = 256
#define WS_BEFF   16640                   // f32 [16][256]      = 16384
#define WS_WEFF   33024                   // bf16 [16][9][256][256] = 18874368

typedef __attribute__((ext_vector_type(16))) __bf16 v16bf;
typedef __attribute__((ext_vector_type(8)))  float  v8f;
typedef __attribute__((ext_vector_type(4)))  unsigned int u32x4;

static __device__ __forceinline__ unsigned short f32_to_bf16(float f) {
    unsigned int u = __float_as_uint(f);
    unsigned int lsb = (u >> 16) & 1u;
    u += 0x7fffu + lsb;                 // round-to-nearest-even
    return (unsigned short)(u >> 16);
}

// ================= kernel 1: global average pool =================
__global__ void pool_kernel(const float* __restrict__ x, float* __restrict__ pooled) {
    const int c = blockIdx.x, b = blockIdx.y;
    const float* p = x + (((size_t)(b * CC + c)) << 12);
    float s = 0.0f;
    for (int i = threadIdx.x; i < HW; i += 256) s += p[i];
    __shared__ float red[256];
    red[threadIdx.x] = s;
    __syncthreads();
    for (int off = 128; off > 0; off >>= 1) {
        if ((int)threadIdx.x < off) red[threadIdx.x] += red[threadIdx.x + off];
        __syncthreads();
    }
    if (threadIdx.x == 0) pooled[b * CC + c] = red[0] * (1.0f / (float)HW);
}

// ================= kernel 2: router (single block) =================
__global__ void router_kernel(const float* __restrict__ pooled,
                              const float* __restrict__ time_emb,
                              const float* __restrict__ Wq, const float* __restrict__ bq,
                              const float* __restrict__ Wk, const float* __restrict__ bk,
                              const float* __restrict__ Wv, const float* __restrict__ bv,
                              const float* __restrict__ Wm1, const float* __restrict__ bm1,
                              const float* __restrict__ Wm2, const float* __restrict__ bm2,
                              const float* __restrict__ Wc, const float* __restrict__ bc,
                              const float* __restrict__ expert_b,
                              float* __restrict__ rw, float* __restrict__ beff) {
    __shared__ float s_xatt[CC];
    __shared__ float s_h1[CC / 4];
    __shared__ float s_xmix[CC];
    __shared__ float s_rw[EE];
    const int t = threadIdx.x;   // one output channel per thread
    for (int b = 0; b < BB; ++b) {
        const float* te = time_emb + b * TDIM;
        const float* pl = pooled + b * CC;
        float q = bq[t], k = bk[t], v = bv[t];
        for (int j = 0; j < TDIM; ++j) q += te[j] * Wq[t * TDIM + j];
        for (int j = 0; j < CC; ++j) {
            const float pj = pl[j];
            k += pj * Wk[t * CC + j];
            v += pj * Wv[t * CC + j];
        }
        const float attn = 1.0f / (1.0f + expf(-(q * k)));
        const float xa = v * attn;
        s_xatt[t] = xa;
        __syncthreads();
        if (t < CC / 4) {
            float m = bm1[t];
            for (int j = 0; j < CC; ++j) m += s_xatt[j] * Wm1[t * CC + j];
            s_h1[t] = 0.5f * m * (1.0f + erff(m * 0.70710678118654752440f)); // exact GELU
        }
        __syncthreads();
        float h2 = bm2[t];
        for (int j = 0; j < CC / 4; ++j) h2 += s_h1[j] * Wm2[t * (CC / 4) + j];
        s_xmix[t] = xa + h2;
        __syncthreads();
        if (t < EE) {
            float lg = bc[t];
            for (int j = 0; j < CC; ++j) lg += s_xmix[j] * Wc[t * CC + j];
            const float r = tanhf(lg);
            s_rw[t] = r;
            rw[b * EE + t] = r;
        }
        __syncthreads();
        float be = 0.0f;
        for (int e = 0; e < EE; ++e) be += s_rw[e] * expert_b[e * OO + t];
        beff[b * OO + t] = be;
        __syncthreads();
    }
}

// ================= kernel 3: expert weight mixing -> bf16 =================
// weff layout: [B][khw][O][C] bf16 (K-minor for direct 128b LDS tile copies).
__global__ void mix_kernel(const float* __restrict__ expert_w,
                           const float* __restrict__ rw,
                           unsigned short* __restrict__ weff) {
    const int g = blockIdx.x * 256 + threadIdx.x;  // < 9*256*256
    const int c = g & 255;
    const int o = (g >> 8) & 255;
    const int khw = g >> 16;
    const size_t base = ((size_t)o * CC + c) * 9 + khw;
    const size_t estride = (size_t)OO * CC * 9;
    const float w0 = expert_w[base];
    const float w1 = expert_w[base + estride];
    const float w2 = expert_w[base + 2 * estride];
    const float w3 = expert_w[base + 3 * estride];
#pragma unroll
    for (int b = 0; b < BB; ++b) {
        const float* r = rw + b * EE;
        const float acc = r[0] * w0 + r[1] * w1 + r[2] * w2 + r[3] * w3;
        weff[((((size_t)b * 9 + khw) * OO + o) << 8) + c] = f32_to_bf16(acc);
    }
}

// ================= kernel 4: implicit-GEMM conv via WMMA bf16 =================
// Block tile: M=64 (output channels) x N=64 (one output row of pixels).
// 8 waves: wm in {0,1} x wn in {0..3}; wave owns M rows {wm*16, wm*16+32}.
// Outer K loop: kh (3 input rows) x c0 (8 chunks of 32 channels).
// All three kw shifts consumed per staging period from ONE padded x-row tile:
//   bT[66][32]: column w' stored at index w'+1; indices 0 and 65 are the
//   always-out-of-image pad columns, zeroed once at kernel start.
// Per wave per period: 6 x v_wmma_f32_16x16x32_bf16.
#define ASTRIDE 40   // ushorts per LDS row (80B: 16B-aligned, conflict-free)
#define BSTRIDE 40

__global__ __launch_bounds__(256) void conv_kernel(const float* __restrict__ x,
                                                   const unsigned short* __restrict__ weff,
                                                   const float* __restrict__ beff,
                                                   float* __restrict__ out) {
    const int b    = blockIdx.z;
    const int o0   = blockIdx.y * 64;    // M tile base
    const int hrow = blockIdx.x;         // N tile = output row hrow

    const int tid  = threadIdx.x;
    const int lane = tid & 31;
    const int wave = tid >> 5;
    const int wm   = wave >> 2;          // 0..1
    const int wn   = wave & 3;           // 0..3

    __shared__ __align__(16) unsigned short aT[3 * 64 * ASTRIDE]; // [kw][o][k]
    __shared__ __align__(16) unsigned short bT[66 * BSTRIDE];     // [w'+1][k]

    v8f acc0 = {0.f, 0.f, 0.f, 0.f, 0.f, 0.f, 0.f, 0.f};
    v8f acc1 = {0.f, 0.f, 0.f, 0.f, 0.f, 0.f, 0.f, 0.f};

    // fragment addressing (ISA 7.12.2, wave32)
    const int mloc  = lane & 15;
    const int aoff  = (lane >= 16) ? 8 : 0;    // A: lane-half K interleave (+8)
    const int boff  = (lane >= 16) ? 8 : 0;    // B: lane-half K interleave (+8)
    const int arow0 = wm * 16 + mloc;
    const int arow1 = arow0 + 32;
    const int brow  = wn * 16 + mloc;          // output pixel column w

    // cooperative-load indices
    const int l_rowA = tid >> 2, l_partA = tid & 3;  // A: 64 rows x 4 chunks of 8 bf16
    const int pix    = tid & 63;                     // B: one pixel ...
    const int grp    = tid >> 6;                     //    ... x 8 channels (grp*8..+7)

    // zero the two permanent pad columns (w' = -1 and w' = 64)
    if (tid < 32) {
        bT[tid] = 0;
        bT[65 * BSTRIDE + tid] = 0;
    }

    union Frag { v16bf v; u32x4 q[2]; };
    const unsigned short* wbase = weff + (((size_t)b * 9) << 16); // [khw][o][c]

    for (int kh = 0; kh < 3; ++kh) {
        const int hp = hrow + kh - 1;
        if (hp < 0 || hp >= HH) continue;      // uniform: padded rows contribute 0
        for (int c0 = 0; c0 < CC; c0 += 32) {
            __syncthreads();   // previous period's fragment reads complete

            // ---- stage A slabs for kw = 0..2: weff[b][kh*3+kw][o0+..][c0+..]
#pragma unroll
            for (int kw = 0; kw < 3; ++kw) {
                const unsigned short* src = wbase
                    + ((((size_t)(kh * 3 + kw)) << 16))
                    + (((size_t)(o0 + l_rowA)) << 8) + c0 + l_partA * 8;
                *(u32x4*)&aT[(kw * 64 + l_rowA) * ASTRIDE + l_partA * 8] =
                    *(const u32x4*)src;
            }
            // ---- stage padded x-row tile: bT[pix+1][grp*8 .. +7]
            {
                const float* xcol =
                    x + ((((size_t)b * CC) + c0 + grp * 8) << 12) + (hp << 6) + pix;
                unsigned short h[8];
#pragma unroll
                for (int i = 0; i < 8; ++i) h[i] = f32_to_bf16(xcol[(size_t)i << 12]);
                u32x4 packed;
                packed.x = ((unsigned int)h[1] << 16) | h[0];
                packed.y = ((unsigned int)h[3] << 16) | h[2];
                packed.z = ((unsigned int)h[5] << 16) | h[4];
                packed.w = ((unsigned int)h[7] << 16) | h[6];
                *(u32x4*)&bT[(pix + 1) * BSTRIDE + grp * 8] = packed;
                if (c0 + 32 < CC)
                    __builtin_prefetch(xcol + ((size_t)32 << 12), 0, 0);
            }
            __syncthreads();

            // ---- 3 kw positions x 2 M-tiles = 6 WMMAs per wave
#pragma unroll
            for (int kw = 0; kw < 3; ++kw) {
                Frag fa0, fa1, fb;
                const int abase = (kw * 64) * ASTRIDE;
                fb.q[0]  = *(const u32x4*)&bT[(brow + kw) * BSTRIDE + boff];
                fb.q[1]  = *(const u32x4*)&bT[(brow + kw) * BSTRIDE + boff + 16];
                fa0.q[0] = *(const u32x4*)&aT[abase + arow0 * ASTRIDE + aoff];
                fa0.q[1] = *(const u32x4*)&aT[abase + arow0 * ASTRIDE + aoff + 16];
                fa1.q[0] = *(const u32x4*)&aT[abase + arow1 * ASTRIDE + aoff];
                fa1.q[1] = *(const u32x4*)&aT[abase + arow1 * ASTRIDE + aoff + 16];
                acc0 = __builtin_amdgcn_wmma_f32_16x16x32_bf16(
                    false, fa0.v, false, fb.v, (short)0, acc0, false, false);
                acc1 = __builtin_amdgcn_wmma_f32_16x16x32_bf16(
                    false, fa1.v, false, fb.v, (short)0, acc1, false, false);
            }
        }
    }

    // ---- epilogue: C/D layout — VGPR r: M=r (lanes 0-15) / r+8 (lanes 16-31)
    const int pcol  = hrow * WW + wn * 16 + mloc;
    const int obase = o0 + wm * 16 + ((lane >= 16) ? 8 : 0);
    const float* bE = beff + b * OO;
#pragma unroll
    for (int r = 0; r < 8; ++r) {
        const int oA = obase + r;
        const int oB = oA + 32;
        out[(((size_t)(b * OO + oA)) << 12) + pcol] = acc0[r] + bE[oA];
        out[(((size_t)(b * OO + oB)) << 12) + pcol] = acc1[r] + bE[oB];
    }
}

// ================= host launcher =================
extern "C" void kernel_launch(void* const* d_in, const int* in_sizes, int n_in,
                              void* d_out, int out_size, void* d_ws, size_t ws_size,
                              hipStream_t stream) {
    const float* x        = (const float*)d_in[0];
    const float* time_emb = (const float*)d_in[1];
    const float* Wq       = (const float*)d_in[2];
    const float* bq       = (const float*)d_in[3];
    const float* Wk       = (const float*)d_in[4];
    const float* bk       = (const float*)d_in[5];
    const float* Wv       = (const float*)d_in[6];
    const float* bv       = (const float*)d_in[7];
    const float* Wm1      = (const float*)d_in[8];
    const float* bm1      = (const float*)d_in[9];
    const float* Wm2      = (const float*)d_in[10];
    const float* bm2      = (const float*)d_in[11];
    const float* Wc       = (const float*)d_in[12];
    const float* bc       = (const float*)d_in[13];
    const float* expert_w = (const float*)d_in[14];
    const float* expert_b = (const float*)d_in[15];

    char* ws = (char*)d_ws;
    float* pooled        = (float*)(ws + WS_POOLED);
    float* rw            = (float*)(ws + WS_RW);
    float* beff          = (float*)(ws + WS_BEFF);
    unsigned short* weff = (unsigned short*)(ws + WS_WEFF);
    float* out           = (float*)d_out;

    pool_kernel<<<dim3(CC, BB), 256, 0, stream>>>(x, pooled);
    router_kernel<<<1, 256, 0, stream>>>(pooled, time_emb, Wq, bq, Wk, bk, Wv, bv,
                                         Wm1, bm1, Wm2, bm2, Wc, bc, expert_b, rw, beff);
    mix_kernel<<<(9 * OO * CC) / 256, 256, 0, stream>>>(expert_w, rw, weff);
    conv_kernel<<<dim3(HH, OO / 64, BB), 256, 0, stream>>>(x, weff, beff, out);
}
